// Entity_Resolution_Head_5308579577965
// MI455X (gfx1250) — compile-verified, matrix operand order checked by reading, not run on hardware
//
#include <hip/hip_runtime.h>
#include <hip/hip_bf16.h>

// ---------------------------------------------------------------------------
// Entity Resolution Head on gfx1250 (CDNA5, wave32, WMMA)
//
// Pipeline (all on `stream`, deterministic, scratch in d_ws):
//   1. convt: convert each weight f32 (K,N) -> f16 transposed (N,K)
//   2. gather: span features (first/last/mean A,B) + pronoun row -> f16
//   3. wmma GEMMs (v_wmma_f32_16x16x32_f16, f32 accum, 2x2 tiles per wave):
//        G1: pron_emb @ Wp1 -> f32   ; LN+leaky -> f16
//        G2: act @ Wp2 -> f16 into xcat[:, 0:512]
//        G3: ent_emb  @ We1 -> f32   ; LN+leaky -> f16
//        G4: act @ We2 -> f16 into xcat[:, 512:1024]
//        G5: xcat @ Wl -> exact-GeLU -> f32
//   4. classifier: y @ Wc + bc -> d_out (256 x 3), wave-level reduction
// ---------------------------------------------------------------------------

typedef _Float16 half_t;
typedef __attribute__((ext_vector_type(16))) half_t v16h;
typedef __attribute__((ext_vector_type(8)))  half_t v8h;
typedef __attribute__((ext_vector_type(8)))  float  v8f;

#define ERH_B   256
#define ERH_S   512
#define ERH_H   1024
#define ERH_HH  512

// ---------------------------------------------------------------------------
// f32 (K,N) row-major -> f16 (N,K) row-major (transposed for WMMA B lanes)
// ---------------------------------------------------------------------------
__global__ __launch_bounds__(256)
void erh_convt_kernel(const float* __restrict__ in, half_t* __restrict__ out,
                      int K, int N) {
  int idx = blockIdx.x * 256 + threadIdx.x;
  if (idx >= K * N) return;
  int k = idx / N;
  int n = idx - k * N;
  out[(size_t)n * K + k] = (half_t)in[idx];
}

// ---------------------------------------------------------------------------
// Span-feature gather: pron_emb (B,H) f16 and ent_emb (B,6H) f16
// ent row = [firstA, lastA, meanA, firstB, lastB, meanB]
// ---------------------------------------------------------------------------
__global__ __launch_bounds__(256)
void erh_gather_kernel(const float* __restrict__ bert,
                       const int* __restrict__ offs,
                       half_t* __restrict__ pron,
                       half_t* __restrict__ ent) {
  int b = blockIdx.x;
  int sA = offs[b * 5 + 0], eA = offs[b * 5 + 1];
  int sB = offs[b * 5 + 2], eB = offs[b * 5 + 3];
  int pr = offs[b * 5 + 4];
  const float* base = bert + (size_t)b * ERH_S * ERH_H;
  half_t* e = ent + (size_t)b * 6 * ERH_H;
  float invA = 1.0f / (float)(eA - sA);
  float invB = 1.0f / (float)(eB - sB);
#pragma unroll
  for (int i = 0; i < 4; ++i) {
    int h = threadIdx.x + 256 * i;
    pron[(size_t)b * ERH_H + h] = (half_t)base[(size_t)pr * ERH_H + h];
    float fA = base[(size_t)sA * ERH_H + h];
    float lA = base[(size_t)(eA - 1) * ERH_H + h];
    float mA = 0.f;
    for (int t = sA; t < eA; ++t) mA += base[(size_t)t * ERH_H + h];
    mA *= invA;
    float fB = base[(size_t)sB * ERH_H + h];
    float lB = base[(size_t)(eB - 1) * ERH_H + h];
    float mB = 0.f;
    for (int t = sB; t < eB; ++t) mB += base[(size_t)t * ERH_H + h];
    mB *= invB;
    e[h]              = (half_t)fA;
    e[ERH_H + h]      = (half_t)lA;
    e[2 * ERH_H + h]  = (half_t)mA;
    e[3 * ERH_H + h]  = (half_t)fB;
    e[4 * ERH_H + h]  = (half_t)lB;
    e[5 * ERH_H + h]  = (half_t)mB;
  }
}

// ---------------------------------------------------------------------------
// WMMA GEMM: C(MxN,f32 accum) = A(MxK f16, row-major) * Bt(NxK f16)^T + bias
// 2x2 register tiling: each wave owns a 32x32 output macro-tile (4 16x16
// accumulators, 32 VGPRs). Each A/B fragment pair feeds two WMMAs ->
// 1 wmma per 2 b128 loads. 8 waves / 256-thread block.
//
// A-lane striping (16-bit A 16x32): lanes 0-15 hold K {0..7,16..23},
// lanes 16-31 hold K {8..15,24..31}; B mirrors with lane = column.
// C/D: VGPR v -> row v (lanes 0-15) / row v+8 (lanes 16-31), col = lane%16.
// mode 0: f32 out      mode 1: f16 out (concat slice)      mode 2: GeLU f32
// ---------------------------------------------------------------------------
__device__ __forceinline__ v16h erh_merge16(v8h lo, v8h hi) {
  return __builtin_shufflevector(lo, hi, 0, 1, 2, 3, 4, 5, 6, 7,
                                 8, 9, 10, 11, 12, 13, 14, 15);
}

__global__ __launch_bounds__(256)
void erh_gemm_wmma_kernel(const half_t* __restrict__ A,
                          const half_t* __restrict__ Bt,
                          const float* __restrict__ bias,
                          float* __restrict__ Cf,
                          half_t* __restrict__ Ch,
                          int M, int N, int K, int ldc, int coloff, int mode) {
  const int wave = threadIdx.x >> 5;
  const int lane = threadIdx.x & 31;
  const int nt2cnt = N >> 5;                 // 32-wide macro-tiles in N
  const int job = blockIdx.x * 8 + wave;
  const int mt2 = job / nt2cnt;
  const int nt2 = job - mt2 * nt2cnt;
  if (mt2 >= (M >> 5)) return;  // wave-uniform

  const int mn  = lane & 15;            // row within A tile / col within B tile
  const int off = (lane >> 4) << 3;     // 0 or 8: K-octet select per half-wave

  const half_t* ap0 = A  + (size_t)(mt2 * 32 + mn) * K + off;
  const half_t* ap1 = ap0 + (size_t)16 * K;
  const half_t* bp0 = Bt + (size_t)(nt2 * 32 + mn) * K + off;
  const half_t* bp1 = bp0 + (size_t)16 * K;

  v8f acc00 = {}, acc01 = {}, acc10 = {}, acc11 = {};
  for (int k = 0; k < K; k += 32) {
    __builtin_prefetch(ap0 + k + 256, 0, 1);
    __builtin_prefetch(ap1 + k + 256, 0, 1);
    __builtin_prefetch(bp0 + k + 256, 0, 1);
    __builtin_prefetch(bp1 + k + 256, 0, 1);
    v16h a0 = erh_merge16(*(const v8h*)(ap0 + k), *(const v8h*)(ap0 + k + 16));
    v16h a1 = erh_merge16(*(const v8h*)(ap1 + k), *(const v8h*)(ap1 + k + 16));
    v16h b0 = erh_merge16(*(const v8h*)(bp0 + k), *(const v8h*)(bp0 + k + 16));
    v16h b1 = erh_merge16(*(const v8h*)(bp1 + k), *(const v8h*)(bp1 + k + 16));
    acc00 = __builtin_amdgcn_wmma_f32_16x16x32_f16(false, a0, false, b0,
                                                   (short)0, acc00, false, false);
    acc01 = __builtin_amdgcn_wmma_f32_16x16x32_f16(false, a0, false, b1,
                                                   (short)0, acc01, false, false);
    acc10 = __builtin_amdgcn_wmma_f32_16x16x32_f16(false, a1, false, b0,
                                                   (short)0, acc10, false, false);
    acc11 = __builtin_amdgcn_wmma_f32_16x16x32_f16(false, a1, false, b1,
                                                   (short)0, acc11, false, false);
  }

  const int col0 = nt2 * 32 + mn;
  const int col1 = col0 + 16;
  const float bv0 = bias[col0];
  const float bv1 = bias[col1];
  const int rbase = mt2 * 32 + off;     // off == ((lane>>4)<<3)

#pragma unroll
  for (int v = 0; v < 8; ++v) {
    const int row0 = rbase + v;
    const int row1 = row0 + 16;
    float x00 = acc00[v] + bv0;
    float x01 = acc01[v] + bv1;
    float x10 = acc10[v] + bv0;
    float x11 = acc11[v] + bv1;
    size_t o00 = (size_t)row0 * ldc + coloff + col0;
    size_t o01 = (size_t)row0 * ldc + coloff + col1;
    size_t o10 = (size_t)row1 * ldc + coloff + col0;
    size_t o11 = (size_t)row1 * ldc + coloff + col1;
    if (mode == 0) {
      Cf[o00] = x00; Cf[o01] = x01; Cf[o10] = x10; Cf[o11] = x11;
    } else if (mode == 1) {
      Ch[o00] = (half_t)x00; Ch[o01] = (half_t)x01;
      Ch[o10] = (half_t)x10; Ch[o11] = (half_t)x11;
    } else {
      const float is2 = 0.70710678118654752f;
      Cf[o00] = 0.5f * x00 * (1.0f + erff(x00 * is2));
      Cf[o01] = 0.5f * x01 * (1.0f + erff(x01 * is2));
      Cf[o10] = 0.5f * x10 * (1.0f + erff(x10 * is2));
      Cf[o11] = 0.5f * x11 * (1.0f + erff(x11 * is2));
    }
  }
}

// ---------------------------------------------------------------------------
// Row LayerNorm + leaky ReLU, f32 in -> f16 out. One 256-thread block / row.
// ---------------------------------------------------------------------------
__global__ __launch_bounds__(256)
void erh_ln_leaky_kernel(const float* __restrict__ X,
                         const float* __restrict__ g,
                         const float* __restrict__ beta,
                         half_t* __restrict__ Y) {
  __shared__ float s_sum[256];
  __shared__ float s_sq[256];
  const int b = blockIdx.x;
  const int tid = threadIdx.x;
  const float* x = X + (size_t)b * ERH_H;
  float v[4];
  float s = 0.f, sq = 0.f;
#pragma unroll
  for (int i = 0; i < 4; ++i) {
    float t = x[tid + 256 * i];
    v[i] = t;
    s += t;
    sq += t * t;
  }
  s_sum[tid] = s;
  s_sq[tid] = sq;
  __syncthreads();
  for (int st = 128; st > 0; st >>= 1) {
    if (tid < st) {
      s_sum[tid] += s_sum[tid + st];
      s_sq[tid] += s_sq[tid + st];
    }
    __syncthreads();
  }
  const float mean = s_sum[0] * (1.0f / ERH_H);
  const float var = s_sq[0] * (1.0f / ERH_H) - mean * mean;
  const float rs = rsqrtf(var + 1e-5f);
#pragma unroll
  for (int i = 0; i < 4; ++i) {
    int h = tid + 256 * i;
    float y = (v[i] - mean) * rs * g[h] + beta[h];
    y = (y > 0.f) ? y : 0.01f * y;
    Y[(size_t)b * ERH_H + h] = (half_t)y;
  }
}

// ---------------------------------------------------------------------------
// Classifier: out(B,3) = Y(B,512) @ Wc(512,3) + bc. One wave per row.
// ---------------------------------------------------------------------------
__global__ __launch_bounds__(256)
void erh_cls_kernel(const float* __restrict__ Yg,
                    const float* __restrict__ Wc,
                    const float* __restrict__ bc,
                    float* __restrict__ out) {
  const int wave = threadIdx.x >> 5;
  const int lane = threadIdx.x & 31;
  const int row = blockIdx.x * 8 + wave;
  const float* y = Yg + (size_t)row * ERH_HH;
  float acc0 = 0.f, acc1 = 0.f, acc2 = 0.f;
  for (int c = lane; c < ERH_HH; c += 32) {
    float yv = y[c];
    acc0 += yv * Wc[c * 3 + 0];
    acc1 += yv * Wc[c * 3 + 1];
    acc2 += yv * Wc[c * 3 + 2];
  }
#pragma unroll
  for (int sft = 16; sft > 0; sft >>= 1) {
    acc0 += __shfl_down(acc0, sft, 32);
    acc1 += __shfl_down(acc1, sft, 32);
    acc2 += __shfl_down(acc2, sft, 32);
  }
  if (lane == 0) {
    out[row * 3 + 0] = acc0 + bc[0];
    out[row * 3 + 1] = acc1 + bc[1];
    out[row * 3 + 2] = acc2 + bc[2];
  }
}

// ---------------------------------------------------------------------------
extern "C" void kernel_launch(void* const* d_in, const int* in_sizes, int n_in,
                              void* d_out, int out_size, void* d_ws,
                              size_t ws_size, hipStream_t stream) {
  (void)in_sizes; (void)n_in; (void)out_size;
  const float* bert  = (const float*)d_in[0];
  const int*   offs  = (const int*)d_in[1];
  const float* Wp1   = (const float*)d_in[2];
  const float* bp1   = (const float*)d_in[3];
  const float* gp    = (const float*)d_in[4];
  const float* betap = (const float*)d_in[5];
  const float* Wp2   = (const float*)d_in[6];
  const float* bp2   = (const float*)d_in[7];
  const float* We1   = (const float*)d_in[8];
  const float* be1   = (const float*)d_in[9];
  const float* ge    = (const float*)d_in[10];
  const float* betae = (const float*)d_in[11];
  const float* We2   = (const float*)d_in[12];
  const float* be2   = (const float*)d_in[13];
  const float* Wl    = (const float*)d_in[14];
  const float* bl    = (const float*)d_in[15];
  const float* Wc    = (const float*)d_in[16];
  const float* bc    = (const float*)d_in[17];

  char* ws = (char*)d_ws;
  size_t off = 0;
  auto alloc = [&](size_t bytes) -> void* {
    void* p = ws + off;
    off += (bytes + 255) & ~(size_t)255;
    return p;
  };

  half_t* Wp1t   = (half_t*)alloc((size_t)1024 * 1024 * 2);  // (N=1024,K=1024)
  half_t* Wp2t   = (half_t*)alloc((size_t)512 * 1024 * 2);   // (N=512, K=1024)
  half_t* We1t   = (half_t*)alloc((size_t)1024 * 6144 * 2);  // (N=1024,K=6144)
  half_t* We2t   = (half_t*)alloc((size_t)512 * 1024 * 2);
  half_t* Wlt    = (half_t*)alloc((size_t)512 * 1024 * 2);
  half_t* pron_h = (half_t*)alloc((size_t)256 * 1024 * 2);
  half_t* ent_h  = (half_t*)alloc((size_t)256 * 6144 * 2);
  float*  t1     = (float*)alloc((size_t)256 * 1024 * 4);    // GEMM f32 (reused)
  half_t* a1     = (half_t*)alloc((size_t)256 * 1024 * 2);   // LN+leaky f16 (reused)
  half_t* xcat   = (half_t*)alloc((size_t)256 * 1024 * 2);   // [x_pron | x_ent]
  float*  ygelu  = (float*)alloc((size_t)256 * 512 * 4);
  if (off > ws_size) return;  // workspace too small; nothing sane to do

  // 1. Weight convert + transpose (f32 KxN -> f16 NxK)
  erh_convt_kernel<<<(1024 * 1024 + 255) / 256, 256, 0, stream>>>(Wp1, Wp1t, 1024, 1024);
  erh_convt_kernel<<<(1024 * 512 + 255) / 256, 256, 0, stream>>>(Wp2, Wp2t, 1024, 512);
  erh_convt_kernel<<<(6144 * 1024 + 255) / 256, 256, 0, stream>>>(We1, We1t, 6144, 1024);
  erh_convt_kernel<<<(1024 * 512 + 255) / 256, 256, 0, stream>>>(We2, We2t, 1024, 512);
  erh_convt_kernel<<<(1024 * 512 + 255) / 256, 256, 0, stream>>>(Wl, Wlt, 1024, 512);

  // 2. Span gather
  erh_gather_kernel<<<256, 256, 0, stream>>>(bert, offs, pron_h, ent_h);

  // 32x32 macro-tiles: jobs = (M/32)*(N/32), 8 wave-jobs per block
  // G1/G3: (256/32)*(1024/32) = 256 jobs -> 32 blocks
  // G2/G4/G5: (256/32)*(512/32) = 128 jobs -> 16 blocks

  // 3. Pron branch: G1 -> LN -> G2 -> xcat[:,0:512]
  erh_gemm_wmma_kernel<<<32, 256, 0, stream>>>(
      pron_h, Wp1t, bp1, t1, nullptr, 256, 1024, 1024, 1024, 0, 0);
  erh_ln_leaky_kernel<<<256, 256, 0, stream>>>(t1, gp, betap, a1);
  erh_gemm_wmma_kernel<<<16, 256, 0, stream>>>(
      a1, Wp2t, bp2, nullptr, xcat, 256, 512, 1024, 1024, 0, 1);

  // 4. Entity branch: G3 -> LN -> G4 -> xcat[:,512:1024]
  erh_gemm_wmma_kernel<<<32, 256, 0, stream>>>(
      ent_h, We1t, be1, t1, nullptr, 256, 1024, 6144, 1024, 0, 0);
  erh_ln_leaky_kernel<<<256, 256, 0, stream>>>(t1, ge, betae, a1);
  erh_gemm_wmma_kernel<<<16, 256, 0, stream>>>(
      a1, We2t, be2, nullptr, xcat, 256, 512, 1024, 1024, 512, 1);

  // 5. G5: xcat @ Wl + bl -> exact GeLU (f32)
  erh_gemm_wmma_kernel<<<16, 256, 0, stream>>>(
      xcat, Wlt, bl, ygelu, nullptr, 256, 512, 1024, 512, 0, 2);

  // 6. Classifier -> d_out (256 x 3, f32)
  erh_cls_kernel<<<256 / 8, 256, 0, stream>>>(ygelu, Wc, bc, (float*)d_out);
}
